// QMeasurement1_71519795413089
// MI455X (gfx1250) — matrix operand order repaired and checked by reading, not compiled, not run on hardware
//
#include <hip/hip_runtime.h>

// ---------------------------------------------------------------------------
// output (4096x128) = [Xr | Xi] (4096x32768) @ [Pr | -Pi]^T (32768x128)
// bf16 WMMA, 3-term split precision (A_hi*B_hi + A_hi*B_lo + A_lo*B_hi).
// B is staged per-workgroup into LDS with CDNA5 async global->LDS copies
// (double buffered), so all 8 waves share one L2 read of each B tile.
// ---------------------------------------------------------------------------

typedef __attribute__((ext_vector_type(16))) __bf16 v16bf;
typedef __attribute__((ext_vector_type(8)))  float  v8f;

static __device__ __forceinline__ __bf16 f2bf_rne(float x) {
    unsigned u = __builtin_bit_cast(unsigned, x);
    unsigned r = (u + 0x7FFFu + ((u >> 16) & 1u)) >> 16;   // round-to-nearest-even
    return __builtin_bit_cast(__bf16, (unsigned short)r);
}
static __device__ __forceinline__ __bf16 f2bf_trunc(float x) {
    unsigned u = __builtin_bit_cast(unsigned, x);
    return __builtin_bit_cast(__bf16, (unsigned short)(u >> 16));
}
static __device__ __forceinline__ float bf_hi_part(float x) {
    unsigned u = __builtin_bit_cast(unsigned, x) & 0xFFFF0000u;
    return __builtin_bit_cast(float, u);
}

// ---------------------------------------------------------------------------
// Kernel 1: build [Pr | -Pi] in bf16 hi/lo planes, pre-swizzled so that the
// GEMM's B fragment is a contiguous 32-byte load per lane.
//   slot id = (kstep*8 + ntile)*32 + lane ; each slot holds 16 bf16.
//   B-fragment layout (contiguous k, not ISA table order): N = ntile*16 + (lane&15),
//   K_local = (lane>>4)*16 + t,  K_global = kstep*32 + K_local.
//   K_global < 16384 -> Pr[n,i,j]  = R[n,i]R[n,j] + I[n,i]I[n,j]
//   else             -> -Pi[n,i,j] = R[n,i]I[n,j] - I[n,i]R[n,j]
// ---------------------------------------------------------------------------
__global__ __launch_bounds__(256) void build_projector(
    const float* __restrict__ ku,      // (128,128,2)
    __bf16* __restrict__ Ph,           // 128*32768 bf16 (swizzled)
    __bf16* __restrict__ Pl)
{
    int id   = blockIdx.x * 256 + threadIdx.x;   // 0 .. 262143
    int lane = id & 31;
    int nt   = (id >> 5) & 7;
    int ks   = id >> 8;                          // global kstep 0..1023
    int n    = nt * 16 + (lane & 15);
    int kb   = ks * 32 + ((lane >> 4) << 4);

    const float* kn = ku + n * 256;              // row n: 128 x {R, I}
    v16bf hi, lo;
    #pragma unroll
    for (int t = 0; t < 16; ++t) {
        int Kg   = kb + t;
        int half = Kg >> 14;                     // 0: Pr, 1: -Pi
        int rem  = Kg & 16383;
        int i = rem >> 7, j = rem & 127;
        float Ri = kn[i * 2 + 0], Ii = kn[i * 2 + 1];
        float Rj = kn[j * 2 + 0], Ij = kn[j * 2 + 1];
        float v  = (half == 0) ? (Ri * Rj + Ii * Ij) : (Ri * Ij - Ii * Rj);
        hi[t] = f2bf_trunc(v);
        lo[t] = f2bf_rne(v - bf_hi_part(v));
    }
    *((v16bf*)Ph + id) = hi;
    *((v16bf*)Pl + id) = lo;
}

// ---------------------------------------------------------------------------
// Kernel 2: split-K WMMA GEMM. grid = (32 M-tiles, 8 K-splits), 8 waves/WG.
// Wave w computes rows [mtile*128 + w*16, +16) x all 128 N columns.
// B staged into LDS: stage = 2 k-steps = 16 KB hi + 16 KB lo, x2 buffers
// = 64 KB dynamic LDS.  Async copies overlap the WMMA phase.
// ---------------------------------------------------------------------------
__global__ __launch_bounds__(256) void wmma_gemm(
    const float* __restrict__ Xr, const float* __restrict__ Xi,
    const __bf16* __restrict__ Ph, const __bf16* __restrict__ Pl,
    float* __restrict__ partial)                 // 8 x 4096 x 128 f32
{
    extern __shared__ char smem[];               // 65536 B: [buf][hi 16K | lo 16K]
    const unsigned smem_base = (unsigned)(uintptr_t)smem;

    const int tid    = threadIdx.x;
    const int lane   = tid & 31;
    const int wave   = tid >> 5;
    const int mtile  = blockIdx.x;               // 0..31
    const int ksplit = blockIdx.y;               // 0..7 (0-3: Xr, 4-7: Xi)
    const int kstep0 = ksplit * 128;             // global kstep base

    // A-fragment addressing (16-bit A 16x32, contiguous k order as in B):
    //   M = lane&15 ; lanes>=16 read K base +16 ; 16 contiguous K per lane.
    const int row = mtile * 128 + wave * 16 + (lane & 15);
    const float* X = (ksplit < 4) ? Xr : Xi;
    const float* arow = X + (size_t)row * 16384
                          + (ksplit & 3) * 4096 + ((lane >> 4) << 4);

    // B staging source: 8192 B per k-step per plane, linear in (kstep,nt,lane).
    const char* PhB = (const char*)Ph + (size_t)kstep0 * 8192;
    const char* PlB = (const char*)Pl + (size_t)kstep0 * 8192;

    auto stage = [&](int o) {                    // stage o covers k-steps [2o, 2o+2)
        const int buf = o & 1;
        unsigned dH = smem_base + buf * 32768 + tid * 16;
        const char* sH = PhB + (size_t)o * 16384 + tid * 16;
        const char* sL = PlB + (size_t)o * 16384 + tid * 16;
        #pragma unroll
        for (int i = 0; i < 4; ++i) {            // 4 KB per slice * 4 = 16 KB/plane
            unsigned dh = dH + i * 4096;
            unsigned dl = dh + 16384;
            asm volatile("global_load_async_to_lds_b128 %0, %1, off"
                         :: "v"(dh),
                            "v"((unsigned long long)(uintptr_t)(sH + i * 4096))
                         : "memory");
            asm volatile("global_load_async_to_lds_b128 %0, %1, off"
                         :: "v"(dl),
                            "v"((unsigned long long)(uintptr_t)(sL + i * 4096))
                         : "memory");
        }
    };

    v8f acc[8];
    #pragma unroll
    for (int n = 0; n < 8; ++n) acc[n] = (v8f){0,0,0,0,0,0,0,0};

    stage(0);
    asm volatile("s_wait_asynccnt 0" ::: "memory");
    __syncthreads();

    for (int o = 0; o < 64; ++o) {               // 64 stages x 2 k-steps = 128 steps
        if (o + 1 < 64) stage(o + 1);            // async; overlaps compute below
        const int buf = o & 1;

        #pragma unroll
        for (int sl = 0; sl < 2; ++sl) {
            const int s = o * 2 + sl;

            // ---- A fragment: 16 f32 -> bf16 hi/lo split in registers ----
            const float* ap = arow + s * 32;
            float4 a0 = *(const float4*)(ap +  0);
            float4 a1 = *(const float4*)(ap +  4);
            float4 a2 = *(const float4*)(ap +  8);
            float4 a3 = *(const float4*)(ap + 12);
            float av[16] = { a0.x, a0.y, a0.z, a0.w,  a1.x, a1.y, a1.z, a1.w,
                             a2.x, a2.y, a2.z, a2.w,  a3.x, a3.y, a3.z, a3.w };
            v16bf ah, al;
            #pragma unroll
            for (int t = 0; t < 16; ++t) {
                ah[t] = f2bf_trunc(av[t]);                     // hi = truncate
                al[t] = f2bf_rne(av[t] - bf_hi_part(av[t]));   // lo = RNE remainder
            }

            // ---- B fragments from LDS (32 B contiguous per lane) ----
            const char* lbase = smem + buf * 32768 + (sl * 8 * 32 + lane) * 32;
            #pragma unroll
            for (int nt = 0; nt < 8; ++nt) {
                v16bf bh = *(const v16bf*)(lbase + nt * 1024);
                v16bf bl = *(const v16bf*)(lbase + nt * 1024 + 16384);
                acc[nt] = __builtin_amdgcn_wmma_f32_16x16x32_bf16(
                    false, ah, false, bh, (short)0, acc[nt], false, false);
                acc[nt] = __builtin_amdgcn_wmma_f32_16x16x32_bf16(
                    false, ah, false, bl, (short)0, acc[nt], false, false);
                acc[nt] = __builtin_amdgcn_wmma_f32_16x16x32_bf16(
                    false, al, false, bh, (short)0, acc[nt], false, false);
            }
        }

        asm volatile("s_wait_asynccnt 0" ::: "memory");  // next buffer written
        __syncthreads();                                 // all waves done reading
    }

    // C/D layout (ISA 7.12.2): VGPR r -> M = r + (lane>=16 ? 8 : 0), N = lane&15.
    float* pout = partial + (size_t)ksplit * (4096 * 128);
    const int mbase = mtile * 128 + wave * 16 + ((lane >> 4) << 3);
    const int ncol  = lane & 15;
    #pragma unroll
    for (int nt = 0; nt < 8; ++nt) {
        #pragma unroll
        for (int r = 0; r < 8; ++r) {
            pout[(size_t)(mbase + r) * 128 + nt * 16 + ncol] = acc[nt][r];
        }
    }
}

// ---------------------------------------------------------------------------
// Kernel 3: deterministic fixed-order reduction of the 8 K-split partials.
// ---------------------------------------------------------------------------
__global__ __launch_bounds__(256) void reduce_partials(
    const float* __restrict__ parts, float* __restrict__ out)
{
    int idx = blockIdx.x * 256 + threadIdx.x;    // 0 .. 524287
    float s = 0.f;
    #pragma unroll
    for (int y = 0; y < 8; ++y) s += parts[(size_t)y * 524288 + idx];
    out[idx] = s;
}

extern "C" void kernel_launch(void* const* d_in, const int* in_sizes, int n_in,
                              void* d_out, int out_size, void* d_ws, size_t ws_size,
                              hipStream_t stream)
{
    const float* Xr = (const float*)d_in[0];     // (4096,128,128)
    const float* Xi = (const float*)d_in[1];     // (4096,128,128)
    const float* ku = (const float*)d_in[2];     // (128,128,2)
    float* out = (float*)d_out;                  // (4096,128)

    // workspace layout: P_hi (8.39 MB) | P_lo (8.39 MB) | partials (16.8 MB)
    __bf16* Ph      = (__bf16*)d_ws;
    __bf16* Pl      = (__bf16*)((char*)d_ws + (size_t)128 * 32768 * 2);
    float*  partial = (float*)((char*)d_ws + (size_t)2 * 128 * 32768 * 2);

    build_projector<<<1024, 256, 0, stream>>>(ku, Ph, Pl);
    wmma_gemm<<<dim3(32, 8), 256, 65536, stream>>>(Xr, Xi, Ph, Pl, partial);
    reduce_partials<<<2048, 256, 0, stream>>>(partial, out);
}
